// DActor_33698313404514
// MI455X (gfx1250) — matrix operand order repaired and checked by reading, not compiled, not run on hardware
//
#include <hip/hip_runtime.h>
#include <cstdint>
#include <cstddef>

#define STATE_DIM 512
#define H1R 500
#define H1P 512
#define H2R 300
#define H2P 320
#define H3R 100
#define H3P 128
#define ACT 64
#define MTILE 128
#define NWAVES 8
#define BATCH_TOTAL 65536

typedef __attribute__((ext_vector_type(16))) __bf16 v16bf;
typedef __attribute__((ext_vector_type(8)))  float  v8f;
typedef __attribute__((ext_vector_type(4)))  unsigned int u32x4;
typedef __attribute__((ext_vector_type(8)))  unsigned int u32x8;

static __device__ __forceinline__ unsigned short f2bf(float f) {
  unsigned int u = __builtin_bit_cast(unsigned int, f);
  u += 0x7fffu + ((u >> 16) & 1u);   // round-to-nearest-even
  return (unsigned short)(u >> 16);
}

// Build a 16 x bf16 WMMA fragment from two 16-byte chunks (8 elements each).
static __device__ __forceinline__ v16bf load_frag(const unsigned short* p0,
                                                  const unsigned short* p1) {
  u32x4 lo = *reinterpret_cast<const u32x4*>(p0);
  u32x4 hi = *reinterpret_cast<const u32x4*>(p1);
  u32x8 w;
  w[0] = lo[0]; w[1] = lo[1]; w[2] = lo[2]; w[3] = lo[3];
  w[4] = hi[0]; w[5] = hi[1]; w[6] = hi[2]; w[7] = hi[3];
  return __builtin_bit_cast(v16bf, w);
}

static __device__ __forceinline__ v8f wmma_bf16(v16bf A, v16bf B, v8f C) {
  return __builtin_amdgcn_wmma_f32_16x16x32_bf16(
      /*neg_a=*/false, A, /*neg_b=*/false, B,
      /*c_mod=*/(short)0, C, /*reuse_a=*/false, /*reuse_b=*/false);
}

// One fused GEMM layer with 2x2 register blocking: each wave owns a 32x32
// output macro-tile, so each k-step is 2 A-frags + 2 B-frags -> 4 WMMAs.
// actIn: LDS bf16 [MTILE][Kpad]; Wt: global bf16 column-major [Npad][Kpad].
// MODE 0: +bias, ReLU -> bf16 LDS out.  MODE 1: +bias, raw f32 LDS out.
template <int MODE>
static __device__ __forceinline__ void gemm_layer(
    const unsigned short* __restrict__ actIn, int Kpad,
    const unsigned short* __restrict__ Wt,
    const float* __restrict__ bias, int Nreal, int Npad,
    unsigned short* __restrict__ outBf, float* __restrict__ outF32,
    int wave, int lane) {
  const int lr   = lane & 15;
  const int half = lane >> 4;
  const int tilesN2 = Npad >> 5;            // 32-wide macro tiles in N
  const int tilesTotal = (MTILE >> 5) * tilesN2;

  for (int t = wave; t < tilesTotal; t += NWAVES) {
    const int mt = t / tilesN2;
    const int nt = t - mt * tilesN2;
    v8f acc00 = {0.f,0.f,0.f,0.f,0.f,0.f,0.f,0.f};
    v8f acc01 = acc00, acc10 = acc00, acc11 = acc00;

    // A rows mt*32 + {0..15} and {16..31}; K pairs at half*8+{0..7}, 16+half*8+{0..7}
    const unsigned short* arow0 = actIn + (mt * 32 + lr) * Kpad + half * 8;
    const unsigned short* arow1 = arow0 + 16 * Kpad;
    // B cols nt*32 + {0..15} and {16..31}; K = half*8+{0..7}, 16+half*8+{0..7}
    const unsigned short* bcol0 = Wt + (size_t)(nt * 32 + lr) * Kpad + half * 8;
    const unsigned short* bcol1 = bcol0 + 16 * Kpad;

    for (int kc = 0; kc < Kpad; kc += 32) {
      v16bf A0 = load_frag(arow0 + kc, arow0 + kc + 16);
      v16bf A1 = load_frag(arow1 + kc, arow1 + kc + 16);
      v16bf B0 = load_frag(bcol0 + kc, bcol0 + kc + 16);
      v16bf B1 = load_frag(bcol1 + kc, bcol1 + kc + 16);
      acc00 = wmma_bf16(A0, B0, acc00);
      acc01 = wmma_bf16(A0, B1, acc01);
      acc10 = wmma_bf16(A1, B0, acc10);
      acc11 = wmma_bf16(A1, B1, acc11);
    }

#pragma unroll
    for (int i = 0; i < 2; ++i) {
#pragma unroll
      for (int j = 0; j < 2; ++j) {
        v8f acc = (i == 0) ? (j == 0 ? acc00 : acc01)
                           : (j == 0 ? acc10 : acc11);
        const int col = nt * 32 + j * 16 + lr;
        const float bv = (col < Nreal) ? bias[col] : 0.f;
#pragma unroll
        for (int r = 0; r < 8; ++r) {
          const int row = mt * 32 + i * 16 + half * 8 + r;  // 32-bit C/D layout
          float v = acc[r] + bv;
          if (MODE == 0) {
            v = fmaxf(v, 0.f);
            outBf[row * Npad + col] = f2bf(v);
          } else {
            outF32[row * ACT + col] = v;
          }
        }
      }
    }
  }
}

extern "C" __global__ void prep_wt(const float* __restrict__ W, int K, int N,
                                   unsigned short* __restrict__ Wt,
                                   int Kpad, int Npad) {
  int idx = blockIdx.x * blockDim.x + threadIdx.x;
  if (idx >= Kpad * Npad) return;
  int n = idx / Kpad;
  int k = idx - n * Kpad;
  float v = (k < K && n < N) ? W[(size_t)k * N + n] : 0.f;
  Wt[idx] = f2bf(v);
}

extern "C" __global__ void __launch_bounds__(256, 1)
dactor_fused(const float* __restrict__ state,
             const unsigned short* __restrict__ Wt1, const float* __restrict__ b1,
             const unsigned short* __restrict__ Wt2, const float* __restrict__ b2,
             const unsigned short* __restrict__ Wt3, const float* __restrict__ b3,
             const unsigned short* __restrict__ Wt4, const float* __restrict__ b4,
             float* __restrict__ out) {
  extern __shared__ unsigned char smem[];
  unsigned short* actA = reinterpret_cast<unsigned short*>(smem);           // 128*512 bf16
  unsigned short* actB = reinterpret_cast<unsigned short*>(smem + 131072);  // 128*512 bf16
  float* outF32 = reinterpret_cast<float*>(smem);                           // 128*64 f32 (reuses actA)

  const int tid  = threadIdx.x;
  const int wave = tid >> 5;
  const int lane = tid & 31;
  const size_t rowBase = (size_t)blockIdx.x * MTILE;

  // Stage state tile -> bf16 LDS (coalesced b128 global loads, b64 LDS stores)
  const float* gstate = state + rowBase * STATE_DIM;
  for (int i = tid * 4; i < MTILE * STATE_DIM; i += 256 * 4) {
    float4 v = *reinterpret_cast<const float4*>(gstate + i);
    unsigned long long pk =
        (unsigned long long)f2bf(v.x)         |
        ((unsigned long long)f2bf(v.y) << 16) |
        ((unsigned long long)f2bf(v.z) << 32) |
        ((unsigned long long)f2bf(v.w) << 48);
    *reinterpret_cast<unsigned long long*>(actA + i) = pk;
  }
  __syncthreads();

  gemm_layer<0>(actA, STATE_DIM, Wt1, b1, H1R, H1P, actB, nullptr, wave, lane);
  __syncthreads();
  gemm_layer<0>(actB, H1P,       Wt2, b2, H2R, H2P, actA, nullptr, wave, lane);
  __syncthreads();
  gemm_layer<0>(actA, H2P,       Wt3, b3, H3R, H3P, actB, nullptr, wave, lane);
  __syncthreads();
  gemm_layer<1>(actB, H3P,       Wt4, b4, ACT, ACT, nullptr, outF32, wave, lane);
  __syncthreads();

  // PositiveSoftmaxTanh: one thread per row
  if (tid < MTILE) {
    const float* rp = outF32 + tid * ACT;
    float* op = out + (rowBase + tid) * ACT;
    float possum = 0.f;
    for (int i = 0; i < ACT - 1; ++i) {
      float tv = tanhf(rp[i]);
      possum += (tv > 0.f) ? tv : 0.f;
    }
    const float denom = (possum == 0.f) ? 1.f : possum;
    for (int i = 0; i < ACT - 1; ++i) {
      float tv = tanhf(rp[i]);
      op[i] = (tv > 0.f) ? (tv / denom) : tv;
    }
    op[ACT - 1] = 1.f / (1.f + expf(-rp[ACT - 1]));
  }
}

extern "C" void kernel_launch(void* const* d_in, const int* in_sizes, int n_in,
                              void* d_out, int out_size, void* d_ws, size_t ws_size,
                              hipStream_t stream) {
  const float* state = (const float*)d_in[0];
  const float* W1 = (const float*)d_in[1]; const float* b1 = (const float*)d_in[2];
  const float* W2 = (const float*)d_in[3]; const float* b2 = (const float*)d_in[4];
  const float* W3 = (const float*)d_in[5]; const float* b3 = (const float*)d_in[6];
  const float* W4 = (const float*)d_in[7]; const float* b4 = (const float*)d_in[8];
  float* out = (float*)d_out;

  // Workspace: transposed bf16 weights (total ~0.93 MB)
  unsigned short* Wt1 = (unsigned short*)d_ws;
  unsigned short* Wt2 = Wt1 + (size_t)H1P * STATE_DIM;  // 512*512
  unsigned short* Wt3 = Wt2 + (size_t)H2P * H1P;        // 320*512
  unsigned short* Wt4 = Wt3 + (size_t)H3P * H2P;        // 128*320

  auto prep = [&](const float* W, int K, int N, unsigned short* Wt,
                  int Kpad, int Npad) {
    int total = Kpad * Npad;
    prep_wt<<<(total + 255) / 256, 256, 0, stream>>>(W, K, N, Wt, Kpad, Npad);
  };
  prep(W1, STATE_DIM, H1R, Wt1, STATE_DIM, H1P);
  prep(W2, H1R,       H2R, Wt2, H1P,       H2P);
  prep(W3, H2R,       H3R, Wt3, H2P,       H3P);
  prep(W4, H3R,       ACT, Wt4, H3P,       ACT);

  const size_t lds_bytes = 262144;  // 2 x 128KB ping-pong activation buffers
  (void)hipFuncSetAttribute(reinterpret_cast<const void*>(dactor_fused),
                            hipFuncAttributeMaxDynamicSharedMemorySize,
                            (int)lds_bytes);
  dactor_fused<<<BATCH_TOTAL / MTILE, 256, lds_bytes, stream>>>(
      state, Wt1, b1, Wt2, b2, Wt3, b3, Wt4, b4, out);
}
